// RVQ_20813411516940
// MI455X (gfx1250) — matrix operand order repaired and checked by reading, not compiled, not run on hardware
//
#include <hip/hip_runtime.h>

// ---- RVQ on MI455X (gfx1250, wave32, WMMA bf16) ----
// Stage GEMM (scores = r·cbT - 0.5||cb||^2) via v_wmma_f32_16x16x32_bf16,
// exact f32 residual updates, argmin via in-register running max + shuffles.

typedef __attribute__((ext_vector_type(16))) __bf16 v16bf;
typedef __attribute__((ext_vector_type(8)))  float  v8f;
typedef __attribute__((ext_vector_type(4)))  float  f4;
typedef __attribute__((ext_vector_type(4)))  __bf16 bf4;

#define NQ      8
#define KCODES  1024
#define DDIM    512
#define NTOK    (16 * 2048)
#define LDS_STRIDE 516   // pad 512->516 dwords: consecutive rows shift 4 banks

// ---------------- Prologue: codebook f32 -> bf16 + 0.5*||cb||^2 ----------------
__global__ __launch_bounds__(128)
void rvq_cvt_codebook(const float* __restrict__ cb,
                      __bf16* __restrict__ cbb,
                      float* __restrict__ hnorm) {
  __shared__ float red[128];
  const int row = blockIdx.x;            // one block per code row (Q*K rows)
  const int t   = threadIdx.x;           // 128 threads x 4 floats = 512 dims
  const float* src = cb + (size_t)row * DDIM + t * 4;
  f4 v = *(const f4*)src;
  bf4 o;
  o[0] = (__bf16)v[0]; o[1] = (__bf16)v[1];
  o[2] = (__bf16)v[2]; o[3] = (__bf16)v[3];
  *(bf4*)(cbb + (size_t)row * DDIM + t * 4) = o;
  red[t] = v[0]*v[0] + v[1]*v[1] + v[2]*v[2] + v[3]*v[3];
  __syncthreads();
  for (int off = 64; off > 0; off >>= 1) {
    if (t < off) red[t] += red[t + off];
    __syncthreads();
  }
  if (t == 0) hnorm[row] = 0.5f * red[0];
}

// ---------------- Main: one wave per 16-token tile ----------------
__global__ __launch_bounds__(32)
void rvq_main(const float* __restrict__ z_e,
              const float* __restrict__ cb,      // f32 codebooks (exact gather)
              const __bf16* __restrict__ cbb,    // bf16 codebooks (WMMA B)
              const float* __restrict__ hnorm,   // 0.5*||cb||^2 per code
              float* __restrict__ out) {
  __shared__ float resid[16][LDS_STRIDE];        // f32 residual, 33 KB

  const int lane = threadIdx.x;                  // 0..31 (wave32)
  const int half = lane >> 4;                    // 0/1: half-wave selector
  const int lcol = lane & 15;                    // column / row-in-tile id
  const int tok0 = blockIdx.x * 16;

  // Load the z_e tile into LDS residual (coalesced b128).
#pragma unroll
  for (int r = 0; r < 16; ++r) {
#pragma unroll
    for (int c = 0; c < 4; ++c) {
      const int d = lane * 4 + c * 128;
      *(f4*)&resid[r][d] = *(const f4*)(z_e + (size_t)(tok0 + r) * DDIM + d);
    }
  }
  __syncthreads();  // cross-lane LDS visibility before A-fragment reads

  for (int s = 0; s < NQ; ++s) {
    // ---- Build 16 A fragments (16x32 bf16) from LDS, held in VGPRs ----
    // 16-bit A layout (matches B): lane m=l&15; VGPR0-7 = K kb..kb+15
    // with kb = kc*32 + (l>>4)*16.
    v16bf a[16];
#pragma unroll
    for (int kc = 0; kc < 16; ++kc) {
      const int db = kc * 32 + half * 16;
      f4 p0 = *(const f4*)&resid[lcol][db];
      f4 p1 = *(const f4*)&resid[lcol][db + 4];
      f4 p2 = *(const f4*)&resid[lcol][db + 8];
      f4 p3 = *(const f4*)&resid[lcol][db + 12];
      v16bf af;
      af[0]=(__bf16)p0[0]; af[1]=(__bf16)p0[1]; af[2]=(__bf16)p0[2]; af[3]=(__bf16)p0[3];
      af[4]=(__bf16)p1[0]; af[5]=(__bf16)p1[1]; af[6]=(__bf16)p1[2]; af[7]=(__bf16)p1[3];
      af[8]=(__bf16)p2[0]; af[9]=(__bf16)p2[1]; af[10]=(__bf16)p2[2]; af[11]=(__bf16)p2[3];
      af[12]=(__bf16)p3[0]; af[13]=(__bf16)p3[1]; af[14]=(__bf16)p3[2]; af[15]=(__bf16)p3[3];
      a[kc] = af;
    }

    float bestv[8];
    int   besti[8];
#pragma unroll
    for (int v = 0; v < 8; ++v) { bestv[v] = -3.0e38f; besti[v] = 0; }

    const __bf16* cbs = cbb + (size_t)s * KCODES * DDIM;
    const float*  hns = hnorm + s * KCODES;
    const float*  cbf = cb + (size_t)s * KCODES * DDIM;

    // ---- Scan all 1024 codes, 16 at a time ----
    for (int kt = 0; kt < KCODES / 16; ++kt) {
      const int code = kt * 16 + lcol;           // this lane's column code
      const float hn = hns[code];                // L2-hot scalar per column

      v8f acc;                                    // C init = -0.5||cb||^2 (per column)
#pragma unroll
      for (int v = 0; v < 8; ++v) acc[v] = -hn;

      // B 32x16 layout used here (contiguous k, as A): lane = column, lanes0-15 K=kc*32..+15,
      // lanes16-31 K=kc*32+16..+31 -> 16 contiguous bf16 of a codebook row.
      const __bf16* bbase = cbs + (size_t)code * DDIM + half * 16;
#pragma unroll
      for (int kc = 0; kc < 16; ++kc) {
        v16bf b = *(const v16bf*)(bbase + kc * 32);
        acc = __builtin_amdgcn_wmma_f32_16x16x32_bf16(
            false, a[kc], false, b, (short)0, acc, false, false);
      }

      // Running per-(row, column-lane) argmax; strict > keeps earliest code.
#pragma unroll
      for (int v = 0; v < 8; ++v) {
        if (acc[v] > bestv[v]) { bestv[v] = acc[v]; besti[v] = code; }
      }
    }

    // ---- Reduce argmax across the 16 columns (stay inside 16-lane halves) ----
#pragma unroll
    for (int v = 0; v < 8; ++v) {
#pragma unroll
      for (int off = 8; off >= 1; off >>= 1) {
        float ov = __shfl_xor(bestv[v], off, 32);
        int   oi = __shfl_xor(besti[v], off, 32);
        if (ov > bestv[v] || (ov == bestv[v] && oi < besti[v])) {
          bestv[v] = ov; besti[v] = oi;
        }
      }
    }

    // ---- Emit indices, exact f32 residual update: r -= cb[idx] ----
#pragma unroll
    for (int r = 0; r < 16; ++r) {
      const int src = (r < 8) ? 0 : 16;          // rows 0-7 live in lanes 0-15
      const int code = __shfl(besti[r & 7], src, 32);
      if (lane == 0) {
        out[(size_t)NTOK * DDIM + (size_t)(tok0 + r) * NQ + s] = (float)code;
      }
      const float* qrow = cbf + (size_t)code * DDIM;
#pragma unroll
      for (int c = 0; c < 4; ++c) {
        const int d = lane * 4 + c * 128;
        f4 qv = *(const f4*)(qrow + d);
        f4 rv = *(const f4*)&resid[r][d];
        rv[0] -= qv[0]; rv[1] -= qv[1]; rv[2] -= qv[2]; rv[3] -= qv[3];
        *(f4*)&resid[r][d] = rv;
      }
    }
    __syncthreads();  // updates visible cross-lane before next stage's A frags
  }

  // ---- final_quantized = z_e - residual (== sum of selected codes) ----
#pragma unroll
  for (int r = 0; r < 16; ++r) {
#pragma unroll
    for (int c = 0; c < 4; ++c) {
      const int d = lane * 4 + c * 128;
      f4 z  = *(const f4*)(z_e + (size_t)(tok0 + r) * DDIM + d);
      f4 rv = *(const f4*)&resid[r][d];
      f4 o;
      o[0] = z[0] - rv[0]; o[1] = z[1] - rv[1];
      o[2] = z[2] - rv[2]; o[3] = z[3] - rv[3];
      *(f4*)(out + (size_t)(tok0 + r) * DDIM + d) = o;
    }
  }
}

extern "C" void kernel_launch(void* const* d_in, const int* in_sizes, int n_in,
                              void* d_out, int out_size, void* d_ws, size_t ws_size,
                              hipStream_t stream) {
  const float* z_e = (const float*)d_in[0];   // [16,2048,512] f32
  const float* cb  = (const float*)d_in[1];   // [8,1024,512]  f32

  // Workspace: bf16 codebook copy (8 MB) + 0.5*||cb||^2 (32 KB).
  __bf16* cbb   = (__bf16*)d_ws;
  float*  hnorm = (float*)((char*)d_ws + (size_t)NQ * KCODES * DDIM * 2);

  rvq_cvt_codebook<<<NQ * KCODES, 128, 0, stream>>>(cb, cbb, hnorm);
  rvq_main<<<NTOK / 16, 32, 0, stream>>>(z_e, cb, cbb, hnorm, (float*)d_out);
}
